// ShiftNet_86921548136943
// MI455X (gfx1250) — matrix operand order repaired and checked by reading, not compiled, not run on hardware
//
#include <hip/hip_runtime.h>
#include <hip/hip_bf16.h>

#define BB   32
#define NPTS 8192
#define G1   128
#define M1   32
#define G2   64
#define M2   16
#define KI   32
#define EMB  27

typedef __attribute__((ext_vector_type(16))) _Float16 v16h;
typedef __attribute__((ext_vector_type(8)))  float    v8f;

#if defined(__has_builtin)
#if __has_builtin(__builtin_amdgcn_tensor_load_to_lds) && __has_builtin(__builtin_amdgcn_s_wait_tensorcnt)
#define HAVE_TDM 1
#endif
#if __has_builtin(__builtin_amdgcn_s_cluster_barrier)
#define HAVE_CLUSTER_BARRIER 1
#endif
#endif

#ifdef HAVE_TDM
typedef __attribute__((ext_vector_type(4))) unsigned int u32x4;
typedef __attribute__((ext_vector_type(8))) int          i32x8;
typedef __attribute__((ext_vector_type(4))) int          i32x4;

// Issue a TDM load of `ndw` dwords from contiguous global memory into LDS.
// D# built per CDNA5 ISA ch.8: group0 {count=1, lds_addr, global_addr, type=2},
// group1 {data_size=4B, tensor_dim0=tile_dim0=stride=ndw, tile_dim1=1}, groups 2/3 zero (2D).
// clang-23 toolchain: 6-arg builtin (extra i32x8 SGPR group before cpol) — zero-filled.
__device__ __forceinline__ void tdm_load_1d(const void* gptr, void* lds_ptr,
                                            unsigned int ndw) {
  unsigned long long ga = (unsigned long long)(uintptr_t)gptr;
  unsigned int lds = (unsigned int)(uintptr_t)lds_ptr;  // generic LDS ptr low 32 = offset
  u32x4 g0;
  g0[0] = 1u;                                             // count=1, user descriptor
  g0[1] = lds;                                            // lds_addr (bytes)
  g0[2] = (unsigned int)(ga & 0xffffffffu);               // global_addr[31:0]
  g0[3] = (unsigned int)((ga >> 32) & 0x01ffffffu) | 0x80000000u;  // addr[56:32], type=2
  i32x8 g1;
  g1[0] = (int)(2u << 16);                                // data_size=4B; mask/flags 0
  g1[1] = (int)((ndw & 0xffffu) << 16);                   // tensor_dim0[15:0] @ bits 63:48
  g1[2] = (int)(((ndw >> 16) & 0xffffu)) | (int)(1u << 16);  // tensor_dim0 hi | tensor_dim1=1
  g1[3] = (int)((ndw & 0xffffu) << 16);                   // tile_dim0 @ bits 127:112
  g1[4] = 1;                                              // tile_dim1=1, tile_dim2=0
  g1[5] = (int)ndw;                                       // tensor_dim0_stride[31:0]
  g1[6] = 0;                                              // stride hi / dim1_stride lo
  g1[7] = 0;
  i32x4 gz4;
  gz4[0] = 0; gz4[1] = 0; gz4[2] = 0; gz4[3] = 0;
  i32x8 gz8;
  gz8[0] = 0; gz8[1] = 0; gz8[2] = 0; gz8[3] = 0;
  gz8[4] = 0; gz8[5] = 0; gz8[6] = 0; gz8[7] = 0;
  __builtin_amdgcn_tensor_load_to_lds(g0, g1, gz4, gz4, gz8, 0);
}
#endif

// ---------------------------------------------------------------------------
// One 16-row tile of a 1x1-conv + eval-BN + relu layer via v_wmma_f32_16x16x32_f16.
// in : LDS f16 [16][in_stride], zero-padded to in_stride (in_stride = 32*kchunks)
// out: LDS f16 [16][out_stride], channels [0,orows) written, rest zero-padded.
// w  : global f32 [orows][kreal] row-major. Whole wave (32 lanes) participates.
// ---------------------------------------------------------------------------
__device__ __forceinline__ void wmma_mlp_layer(
    const _Float16* __restrict__ in, int in_stride, int kreal,
    _Float16* __restrict__ out, int out_stride,
    const float* __restrict__ w, const float* __restrict__ bias,
    const float* __restrict__ gmm, const float* __restrict__ beta,
    int orows, bool bn, int lane)
{
  const float BN_SCALE = 0.9999950000375f;  // 1/sqrt(1+1e-5)
  int col   = lane & 15;
  int half_ = lane >> 4;
  int kchunks = (kreal + 31) >> 5;
  int ntiles  = (orows + 15) >> 4;
  for (int nt = 0; nt < ntiles; ++nt) {
    v8f acc = {};
    int ch = nt * 16 + col;
    for (int kc = 0; kc < kchunks; ++kc) {
      int kbase = kc << 5;
      v16h a, bf;
#pragma unroll
      for (int j = 0; j < 16; ++j) {
        // A: 16-bit 16x32 layout — lane holds row (lane&15); K interleave per ISA table.
        int ka = kbase + ((j < 8) ? (j + 8 * half_) : (j + 8 + 8 * half_));
        a[j] = in[col * in_stride + ka];
        // B: 32x16, lane holds column (lane&15); K interleave matches A.
        int kb = kbase + ((j < 8) ? (j + 8 * half_) : (j + 8 + 8 * half_));
        float wv = (ch < orows && kb < kreal) ? w[ch * kreal + kb] : 0.0f;
        bf[j] = (_Float16)wv;
      }
      acc = __builtin_amdgcn_wmma_f32_16x16x32_f16(false, a, false, bf,
                                                   (short)0, acc, false, false);
    }
    if (ch < orows) {
      float bb = bias[ch];
      float gg = bn ? gmm[ch] * BN_SCALE : 1.0f;
      float ee = bn ? beta[ch] : 0.0f;
#pragma unroll
      for (int v = 0; v < 8; ++v) {
        int r = v + 8 * half_;                    // D: M = v + 8*half, N = lane&15
        float val = (acc[v] + bb) * gg + ee;
        out[r * out_stride + ch] = (_Float16)(val > 0.0f ? val : 0.0f);
      }
    }
  }
  for (int c2 = orows + lane; c2 < out_stride; c2 += 32)
    for (int r = 0; r < 16; ++r) out[r * out_stride + c2] = (_Float16)0.0f;
}

// ---------------------------------------------------------------------------
// Positional embedding: [B,N,3] -> [B,N,27]  (x, sin(2^l x), cos(2^l x))
// ---------------------------------------------------------------------------
__global__ void k_pos_embed(const float* __restrict__ x, float* __restrict__ feat) {
#ifdef HAVE_CLUSTER_BARRIER
  __builtin_amdgcn_s_cluster_barrier();  // NOP when ClusterID==0 (not in a cluster)
#endif
  int i = blockIdx.x * blockDim.x + threadIdx.x;
  if (i >= BB * NPTS) return;
  float p0 = x[(size_t)i * 3 + 0];
  float p1 = x[(size_t)i * 3 + 1];
  float p2 = x[(size_t)i * 3 + 2];
  float* o = feat + (size_t)i * EMB;
  o[0] = p0; o[1] = p1; o[2] = p2;
#pragma unroll
  for (int l = 0; l < 4; ++l) {
    float f = (float)(1 << l);
    o[3 + 6 * l + 0] = sinf(f * p0);
    o[3 + 6 * l + 1] = sinf(f * p1);
    o[3 + 6 * l + 2] = sinf(f * p2);
    o[3 + 6 * l + 3] = cosf(f * p0);
    o[3 + 6 * l + 4] = cosf(f * p1);
    o[3 + 6 * l + 5] = cosf(f * p2);
  }
}

// ---------------------------------------------------------------------------
// Deterministic farthest point sampling. One block per batch.
// PPT points per thread kept in registers; block-wide argmax (first-index tie).
// ---------------------------------------------------------------------------
template <int PPT>
__global__ void k_fps(const float* __restrict__ pts, int npts, int niter,
                      float* __restrict__ centers) {
  __shared__ float sval[1024];
  __shared__ int   sidx[1024];
  int b = blockIdx.x, t = threadIdx.x, nth = blockDim.x;
  const float* base = pts + (size_t)b * npts * 3;
  float px[PPT], py[PPT], pz[PPT], dm[PPT];
#pragma unroll
  for (int j = 0; j < PPT; ++j) {
    int i = t * PPT + j;
    px[j] = base[i * 3 + 0];
    py[j] = base[i * 3 + 1];
    pz[j] = base[i * 3 + 2];
    dm[j] = 1e10f;
  }
  int far = 0;
  for (int it = 0; it < niter; ++it) {
    float cx = base[far * 3 + 0], cy = base[far * 3 + 1], cz = base[far * 3 + 2];
    if (t == 0) {
      float* o = centers + ((size_t)b * niter + it) * 3;
      o[0] = cx; o[1] = cy; o[2] = cz;
    }
    float bv = -1.0f; int bi = 0;
#pragma unroll
    for (int j = 0; j < PPT; ++j) {
      float dx = px[j] - cx, dy = py[j] - cy, dz = pz[j] - cz;
      float d = dx * dx + dy * dy + dz * dz;
      dm[j] = fminf(dm[j], d);
      if (dm[j] > bv) { bv = dm[j]; bi = t * PPT + j; }
    }
    sval[t] = bv; sidx[t] = bi;
    __syncthreads();
    for (int s = nth >> 1; s > 0; s >>= 1) {
      if (t < s) {
        float v2 = sval[t + s]; int i2 = sidx[t + s];
        if (v2 > sval[t] || (v2 == sval[t] && i2 < sidx[t])) { sval[t] = v2; sidx[t] = i2; }
      }
      __syncthreads();
    }
    far = sidx[0];
    __syncthreads();
  }
}

// ---------------------------------------------------------------------------
// kNN: for each (b,g) center find Kn nearest of Sn points (ascending, first-index tie).
// Distances staged in dynamic LDS; iterative argmin with mask-out.
// ---------------------------------------------------------------------------
__global__ void k_knn(const float* __restrict__ cent, const float* __restrict__ pts,
                      int Gn, int Sn, int Kn, int* __restrict__ oidx) {
  extern __shared__ float smem[];
  float* d  = smem;
  float* rv = smem + Sn;
  int*   ri = (int*)(rv + blockDim.x);
  int blk = blockIdx.x, b = blk / Gn, g = blk % Gn;
  int t = threadIdx.x, T = blockDim.x;
  const float* c = cent + ((size_t)b * Gn + g) * 3;
  float cx = c[0], cy = c[1], cz = c[2];
  const float* P = pts + (size_t)b * Sn * 3;
  for (int i = t; i < Sn; i += T) {
    float dx = P[i * 3] - cx, dy = P[i * 3 + 1] - cy, dz = P[i * 3 + 2] - cz;
    d[i] = dx * dx + dy * dy + dz * dz;
  }
  __syncthreads();
  int* op = oidx + ((size_t)b * Gn + g) * Kn;
  for (int k = 0; k < Kn; ++k) {
    float bv = 3e38f; int bi = 0;
    for (int i = t; i < Sn; i += T) {
      float v = d[i];
      if (v < bv) { bv = v; bi = i; }
    }
    rv[t] = bv; ri[t] = bi;
    __syncthreads();
    for (int s = T >> 1; s > 0; s >>= 1) {
      if (t < s) {
        float v2 = rv[t + s]; int i2 = ri[t + s];
        if (v2 < rv[t] || (v2 == rv[t] && i2 < ri[t])) { rv[t] = v2; ri[t] = i2; }
      }
      __syncthreads();
    }
    if (t == 0) { op[k] = ri[0]; d[ri[0]] = 3e38f; }
    __syncthreads();
  }
}

// ---------------------------------------------------------------------------
// SA1: gather 32 neighbors' 27-d embed, MLP 27->64->32->64 (WMMA), max over group.
// One block (2 waves) per group; each wave owns a 16-row tile.
// ---------------------------------------------------------------------------
__global__ void k_sa1(const float* __restrict__ feat, const int* __restrict__ knn,
                      const float* w1, const float* b1, const float* g1, const float* e1,
                      const float* w2, const float* b2, const float* g2, const float* e2,
                      const float* w3, const float* b3, const float* g3, const float* e3,
                      float* __restrict__ c1f) {
  __shared__ _Float16 a0[32 * 32];
  __shared__ _Float16 a1[32 * 64];
  __shared__ _Float16 a2[32 * 32];
  __shared__ _Float16 a3[32 * 64];
  int blk = blockIdx.x, b = blk / G1, g = blk % G1, t = threadIdx.x;
  const int* nb = knn + ((size_t)b * G1 + g) * M1;
  __builtin_prefetch(w1, 0, 0);
  for (int e = t; e < 32 * 32; e += 64) {
    int r = e >> 5, k = e & 31;
    float v = (k < EMB) ? feat[((size_t)b * NPTS + nb[r]) * EMB + k] : 0.0f;
    a0[e] = (_Float16)v;
  }
  __syncthreads();
  int wave = t >> 5, lane = t & 31;
  wmma_mlp_layer(a0 + wave * 16 * 32, 32, EMB, a1 + wave * 16 * 64, 64,
                 w1, b1, g1, e1, 64, true, lane);
  wmma_mlp_layer(a1 + wave * 16 * 64, 64, 64, a2 + wave * 16 * 32, 32,
                 w2, b2, g2, e2, 32, true, lane);
  wmma_mlp_layer(a2 + wave * 16 * 32, 32, 32, a3 + wave * 16 * 64, 64,
                 w3, b3, g3, e3, 64, true, lane);
  __syncthreads();
  // max over the 32 group points; 64 threads <-> 64 channels
  float mx = 0.0f;  // post-relu values are >= 0
  for (int r = 0; r < 32; ++r) mx = fmaxf(mx, (float)a3[r * 64 + t]);
  c1f[((size_t)b * G1 + g) * 64 + t] = mx;
}

// ---------------------------------------------------------------------------
// SA2: gather 16 neighbors' 64-d features, MLP 64->64->32->8 (WMMA), max over group.
// One wave per group.
// ---------------------------------------------------------------------------
__global__ void k_sa2(const float* __restrict__ c1f, const int* __restrict__ knn2,
                      const float* w1, const float* b1, const float* g1, const float* e1,
                      const float* w2, const float* b2, const float* g2, const float* e2,
                      const float* w3, const float* b3, const float* g3, const float* e3,
                      float* __restrict__ c2f) {
  __shared__ _Float16 a0[16 * 64];
  __shared__ _Float16 a1[16 * 64];
  __shared__ _Float16 a2[16 * 32];
  __shared__ _Float16 a3[16 * 16];
  int blk = blockIdx.x, b = blk / G2, g = blk % G2, lane = threadIdx.x;
  const int* nb = knn2 + ((size_t)b * G2 + g) * M2;
  for (int e = lane; e < 16 * 64; e += 32) {
    int r = e >> 6, k = e & 63;
    a0[e] = (_Float16)c1f[((size_t)b * G1 + nb[r]) * 64 + k];
  }
  __syncthreads();
  wmma_mlp_layer(a0, 64, 64, a1, 64, w1, b1, g1, e1, 64, true, lane);
  wmma_mlp_layer(a1, 64, 64, a2, 32, w2, b2, g2, e2, 32, true, lane);
  wmma_mlp_layer(a2, 32, 32, a3, 16, w3, b3, g3, e3, 8, true, lane);
  __syncthreads();
  if (lane < 8) {
    float mx = 0.0f;
    for (int r = 0; r < 16; ++r) mx = fmaxf(mx, (float)a3[r * 16 + lane]);
    c2f[((size_t)b * G2 + g) * 8 + lane] = mx;
  }
}

// ---------------------------------------------------------------------------
// FP2: for each c1 center, interp c2f over 32 nearest of 64 c2 centers,
// concat [c1f(64), interp(8)] -> MLP 72->64->32 (WMMA). One wave / 16 centers.
// ---------------------------------------------------------------------------
__global__ void k_fp2(const float* __restrict__ c1, const float* __restrict__ c2,
                      const float* __restrict__ c1f, const float* __restrict__ c2f,
                      const float* w1, const float* b1, const float* g1, const float* e1,
                      const float* w2, const float* b2, const float* g2, const float* e2,
                      float* __restrict__ c1f2) {
  __shared__ float    dloc[16 * G2];
  __shared__ _Float16 a0[16 * 96];
  __shared__ _Float16 a1[16 * 64];
  __shared__ _Float16 a2[16 * 32];
  int blk = blockIdx.x, b = blk / (G1 / 16), tile = blk % (G1 / 16);
  int lane = threadIdx.x;
  if (lane < 16) {
    int i = tile * 16 + lane;
    const float* p = c1 + ((size_t)b * G1 + i) * 3;
    float px = p[0], py = p[1], pz = p[2];
    for (int s = 0; s < G2; ++s) {
      const float* q = c2 + ((size_t)b * G2 + s) * 3;
      float dx = px - q[0], dy = py - q[1], dz = pz - q[2];
      dloc[lane * G2 + s] = dx * dx + dy * dy + dz * dz;
    }
    float interp[8] = {0, 0, 0, 0, 0, 0, 0, 0};
    float wsum = 0.0f;
    for (int k = 0; k < KI; ++k) {
      float bv = 3e38f; int bi = 0;
      for (int s = 0; s < G2; ++s) {
        float v = dloc[lane * G2 + s];
        if (v < bv) { bv = v; bi = s; }
      }
      dloc[lane * G2 + bi] = 3e38f;
      float rc = 1.0f / (bv + 1e-4f);
      wsum += rc;
      const float* f = c2f + ((size_t)b * G2 + bi) * 8;
#pragma unroll
      for (int c = 0; c < 8; ++c) interp[c] += rc * f[c];
    }
    float inv = 1.0f / wsum;
    const float* f1 = c1f + ((size_t)b * G1 + i) * 64;
    for (int k = 0; k < 64; ++k) a0[lane * 96 + k] = (_Float16)f1[k];
#pragma unroll
    for (int c = 0; c < 8; ++c) a0[lane * 96 + 64 + c] = (_Float16)(interp[c] * inv);
    for (int k = 72; k < 96; ++k) a0[lane * 96 + k] = (_Float16)0.0f;
  }
  __syncthreads();
  wmma_mlp_layer(a0, 96, 72, a1, 64, w1, b1, g1, e1, 64, true, lane);
  wmma_mlp_layer(a1, 64, 64, a2, 32, w2, b2, g2, e2, 32, true, lane);
  __syncthreads();
  for (int e = lane; e < 16 * 32; e += 32) {
    int r = e >> 5, c = e & 31;
    c1f2[((size_t)b * G1 + tile * 16 + r) * 32 + c] = (float)a2[e];
  }
}

// ---------------------------------------------------------------------------
// FP1 + head, fused: per point interp c1f2 over 32 nearest of 128 c1 centers,
// concat [embed(27), interp(32)] -> MLP 59->32->32 -> lin 32->64 (relu) -> 64->3,
// out = x + 0.1*shift*x. One wave per 16-point tile.
// The per-batch c1f2 slice (128x32 f32, 16KB contiguous) is staged into LDS via
// the Tensor Data Mover (TENSOR_LOAD_TO_LDS + s_wait_tensorcnt), with a manual
// fill afterwards as a correctness hedge while TDM arg semantics are unverified.
// ---------------------------------------------------------------------------
__global__ void k_fp1_head(const float* __restrict__ x, const float* __restrict__ feat,
                           const float* __restrict__ c1, const float* __restrict__ c1f2,
                           const float* w1, const float* b1, const float* g1, const float* e1,
                           const float* w2, const float* b2, const float* g2, const float* e2,
                           const float* lw1, const float* lb1,
                           const float* lw2, const float* lb2,
                           float* __restrict__ out) {
  __shared__ float    sf2[G1 * 32];
  __shared__ float    sc1[G1 * 3];
  __shared__ float    dloc[16 * G1];
  __shared__ _Float16 a0[16 * 64];
  __shared__ _Float16 a1[16 * 32];
  __shared__ _Float16 a2[16 * 32];
  __shared__ _Float16 a3[16 * 64];
  int blk = blockIdx.x, b = blk / (NPTS / 16), tile = blk % (NPTS / 16);
  int lane = threadIdx.x;
#ifdef HAVE_TDM
  // Async DMA of the whole per-batch c1f2 slice into LDS (single wave per block).
  tdm_load_1d(c1f2 + (size_t)b * G1 * 32, (void*)sf2, G1 * 32);
  __builtin_amdgcn_s_wait_tensorcnt(0);
#endif
  __builtin_prefetch(c1f2 + (size_t)b * G1 * 32, 0, 0);
  for (int e = lane; e < G1 * 3; e += 32)  sc1[e] = c1[(size_t)b * G1 * 3 + e];
  for (int e = lane; e < G1 * 32; e += 32) sf2[e] = c1f2[(size_t)b * G1 * 32 + e];
  __syncthreads();
  if (lane < 16) {
    int n = tile * 16 + lane;
    const float* p = x + ((size_t)b * NPTS + n) * 3;
    float px = p[0], py = p[1], pz = p[2];
    for (int s = 0; s < G1; ++s) {
      float dx = px - sc1[s * 3], dy = py - sc1[s * 3 + 1], dz = pz - sc1[s * 3 + 2];
      dloc[lane * G1 + s] = dx * dx + dy * dy + dz * dz;
    }
    float interp[32];
#pragma unroll
    for (int c = 0; c < 32; ++c) interp[c] = 0.0f;
    float wsum = 0.0f;
    for (int k = 0; k < KI; ++k) {
      float bv = 3e38f; int bi = 0;
      for (int s = 0; s < G1; ++s) {
        float v = dloc[lane * G1 + s];
        if (v < bv) { bv = v; bi = s; }
      }
      dloc[lane * G1 + bi] = 3e38f;
      float rc = 1.0f / (bv + 1e-4f);
      wsum += rc;
#pragma unroll
      for (int c = 0; c < 32; ++c) interp[c] += rc * sf2[bi * 32 + c];
    }
    float inv = 1.0f / wsum;
    const float* fp = feat + ((size_t)b * NPTS + n) * EMB;
    for (int k = 0; k < EMB; ++k) a0[lane * 64 + k] = (_Float16)fp[k];
#pragma unroll
    for (int c = 0; c < 32; ++c) a0[lane * 64 + EMB + c] = (_Float16)(interp[c] * inv);
    for (int k = 59; k < 64; ++k) a0[lane * 64 + k] = (_Float16)0.0f;
  }
  __syncthreads();
  wmma_mlp_layer(a0, 64, 59, a1, 32, w1, b1, g1, e1, 32, true, lane);
  wmma_mlp_layer(a1, 32, 32, a2, 32, w2, b2, g2, e2, 32, true, lane);
  wmma_mlp_layer(a2, 32, 32, a3, 64, lw1, lb1, nullptr, nullptr, 64, false, lane);
  __syncthreads();
  if (lane < 16) {
    int n = tile * 16 + lane;
    float sh[3];
#pragma unroll
    for (int c = 0; c < 3; ++c) {
      float s = lb2[c];
      for (int k = 0; k < 64; ++k) s += (float)a3[lane * 64 + k] * lw2[c * 64 + k];
      sh[c] = s;
    }
    const float* p = x + ((size_t)b * NPTS + n) * 3;
    float* o = out + ((size_t)b * NPTS + n) * 3;
#pragma unroll
    for (int c = 0; c < 3; ++c) o[c] = p[c] + sh[c] * 0.1f * p[c];
  }
}

// ---------------------------------------------------------------------------
extern "C" void kernel_launch(void* const* d_in, const int* in_sizes, int n_in,
                              void* d_out, int out_size, void* d_ws, size_t ws_size,
                              hipStream_t stream) {
  (void)in_sizes; (void)n_in; (void)out_size; (void)ws_size;
  const float* X = (const float*)d_in[0];
  // sa1: layers (w,b,g,be) at 1..12
  const float *s1w1 = (const float*)d_in[1],  *s1b1 = (const float*)d_in[2],
              *s1g1 = (const float*)d_in[3],  *s1e1 = (const float*)d_in[4];
  const float *s1w2 = (const float*)d_in[5],  *s1b2 = (const float*)d_in[6],
              *s1g2 = (const float*)d_in[7],  *s1e2 = (const float*)d_in[8];
  const float *s1w3 = (const float*)d_in[9],  *s1b3 = (const float*)d_in[10],
              *s1g3 = (const float*)d_in[11], *s1e3 = (const float*)d_in[12];
  // sa2: 13..24
  const float *s2w1 = (const float*)d_in[13], *s2b1 = (const float*)d_in[14],
              *s2g1 = (const float*)d_in[15], *s2e1 = (const float*)d_in[16];
  const float *s2w2 = (const float*)d_in[17], *s2b2 = (const float*)d_in[18],
              *s2g2 = (const float*)d_in[19], *s2e2 = (const float*)d_in[20];
  const float *s2w3 = (const float*)d_in[21], *s2b3 = (const float*)d_in[22],
              *s2g3 = (const float*)d_in[23], *s2e3 = (const float*)d_in[24];
  // fp2: 25..32
  const float *f2w1 = (const float*)d_in[25], *f2b1 = (const float*)d_in[26],
              *f2g1 = (const float*)d_in[27], *f2e1 = (const float*)d_in[28];
  const float *f2w2 = (const float*)d_in[29], *f2b2 = (const float*)d_in[30],
              *f2g2 = (const float*)d_in[31], *f2e2 = (const float*)d_in[32];
  // fp1: 33..40
  const float *f1w1 = (const float*)d_in[33], *f1b1 = (const float*)d_in[34],
              *f1g1 = (const float*)d_in[35], *f1e1 = (const float*)d_in[36];
  const float *f1w2 = (const float*)d_in[37], *f1b2 = (const float*)d_in[38],
              *f1g2 = (const float*)d_in[39], *f1e2 = (const float*)d_in[40];
  // head: 41..44
  const float *lw1 = (const float*)d_in[41], *lb1 = (const float*)d_in[42];
  const float *lw2 = (const float*)d_in[43], *lb2 = (const float*)d_in[44];

  float* out = (float*)d_out;

  // Workspace carve-up
  float* feat27 = (float*)d_ws;                         // B*N*27
  float* c1v    = feat27 + (size_t)BB * NPTS * EMB;     // B*128*3
  float* c1f    = c1v    + (size_t)BB * G1 * 3;         // B*128*64
  float* c2v    = c1f    + (size_t)BB * G1 * 64;        // B*64*3
  float* c2f    = c2v    + (size_t)BB * G2 * 3;         // B*64*8
  float* c1f2   = c2f    + (size_t)BB * G2 * 8;         // B*128*32
  int*   knn1   = (int*)(c1f2 + (size_t)BB * G1 * 32);  // B*128*32
  int*   knn2   = knn1 + (size_t)BB * G1 * M1;          // B*64*16

  // 1) positional embedding
  k_pos_embed<<<(BB * NPTS + 255) / 256, 256, 0, stream>>>(X, feat27);
  // 2) FPS on xyz -> c1 (128 centers/batch)
  k_fps<8><<<BB, 1024, 0, stream>>>(X, NPTS, G1, c1v);
  // 3) kNN(xyz -> c1), k=32
  k_knn<<<BB * G1, 256, (size_t)NPTS * 4 + 256 * 8, stream>>>(c1v, X, G1, NPTS, M1, knn1);
  // 4) SA1 MLP + group max -> c1f
  k_sa1<<<BB * G1, 64, 0, stream>>>(feat27, knn1,
                                    s1w1, s1b1, s1g1, s1e1,
                                    s1w2, s1b2, s1g2, s1e2,
                                    s1w3, s1b3, s1g3, s1e3, c1f);
  // 5) FPS on c1 -> c2 (64 centers/batch)
  k_fps<1><<<BB, 128, 0, stream>>>(c1v, G1, G2, c2v);
  // 6) kNN(c1 -> c2), k=16
  k_knn<<<BB * G2, 128, (size_t)G1 * 4 + 128 * 8, stream>>>(c2v, c1v, G2, G1, M2, knn2);
  // 7) SA2 MLP + group max -> c2f
  k_sa2<<<BB * G2, 32, 0, stream>>>(c1f, knn2,
                                    s2w1, s2b1, s2g1, s2e1,
                                    s2w2, s2b2, s2g2, s2e2,
                                    s2w3, s2b3, s2g3, s2e3, c2f);
  // 8) FP2 interp + MLP -> c1f2
  k_fp2<<<BB * (G1 / 16), 32, 0, stream>>>(c1v, c2v, c1f, c2f,
                                           f2w1, f2b1, f2g1, f2e1,
                                           f2w2, f2b2, f2g2, f2e2, c1f2);
  // 9) FP1 interp + MLP + head, fused -> out
  k_fp1_head<<<BB * (NPTS / 16), 32, 0, stream>>>(X, feat27, c1v, c1f2,
                                                  f1w1, f1b1, f1g1, f1e1,
                                                  f1w2, f1b2, f1g2, f1e2,
                                                  lw1, lb1, lw2, lb2, out);
}
